// GNNEncoder_77068893159552
// MI455X (gfx1250) — matrix operand order repaired and checked by reading, not compiled, not run on hardware
//
#include <hip/hip_runtime.h>
#include <hip/hip_bf16.h>
#include <stdint.h>

// ---------------------------------------------------------------------------
// Problem constants (from reference)
// ---------------------------------------------------------------------------
#define NPG        200
#define DEG        16
#define NS_NODES   80000
#define NQ_NODES   80000
#define NT_NODES   160000        // supports ++ queries
#define F_DIM      128
#define H_DIM      128
#define OUT_DIM    64
#define N_GRAPHS   800           // 400 support + 400 query graphs
#define B_EP       16
#define N_WAY      5
#define K_SHOT     5
#define Q_PER      5
#define NEG_SLOPE  0.2f

typedef __attribute__((ext_vector_type(16))) __bf16 v16bf;
typedef __attribute__((ext_vector_type(8)))  float  v8f;

#if defined(__has_builtin)
#  if __has_builtin(__builtin_amdgcn_cvt_pk_bf16_f32)
#    define HAVE_CVT_PK_BF16 1
#  endif
#endif

// ---------------------------------------------------------------------------
// fp32 pair -> packed bf16x2 (round to nearest even)
// ---------------------------------------------------------------------------
__device__ __forceinline__ uint16_t f32_to_bf16(float f) {
    uint32_t u = __float_as_uint(f);
    uint32_t r = u + 0x7FFFu + ((u >> 16) & 1u);
    return (uint16_t)(r >> 16);
}
__device__ __forceinline__ uint32_t pack_bf16x2(float lo, float hi) {
#ifdef HAVE_CVT_PK_BF16
    typedef __attribute__((ext_vector_type(2))) __bf16 v2bf;
    v2bf r = __builtin_amdgcn_cvt_pk_bf16_f32(lo, hi);
    return __builtin_bit_cast(uint32_t, r);
#else
    return (uint32_t)f32_to_bf16(lo) | ((uint32_t)f32_to_bf16(hi) << 16);
#endif
}

// ---------------------------------------------------------------------------
// WMMA GEMM:  out[rows x NCOLS] = A[rows x 128] @ W[128 x NCOLS]  (+bias,relu)
// Optional fused epilogue: es[m] = sum_n out*a_src[n], ed[m] = sum_n out*a_dst[n]
// Block: 256 threads = 8 waves; block tile = 128 rows x NCOLS (full width).
// Wave w owns rows [16w,16w+16) across all NCOLS/16 column tiles.
// Weights staged in LDS as bf16, transposed [n][k], k-stride padded to 134.
// ---------------------------------------------------------------------------
template <int NCOLS>
__global__ __launch_bounds__(256)
void gemm_bf16_wmma(const float* __restrict__ A,      // [rows,128]
                    const float* __restrict__ W,      // [128,NCOLS]
                    const float* __restrict__ bias,   // [NCOLS] or null
                    const float* __restrict__ a_src,  // [NCOLS] or null
                    const float* __restrict__ a_dst,  // [NCOLS] or null
                    float* __restrict__ out,          // [rows,NCOLS]
                    float* __restrict__ es,           // [rows] or null
                    float* __restrict__ ed,           // [rows] or null
                    int do_relu)
{
    constexpr int KP = 134;                 // padded k-stride (ushorts)
    constexpr int NT_T = NCOLS / 16;        // column tiles per wave
    __shared__ uint16_t Blds[NCOLS * KP];   // transposed bf16 weights [n][k]

    const int tid  = threadIdx.x;
    const int wave = tid >> 5;
    const int lane = tid & 31;
    const int half = lane >> 4;
    const int l16  = lane & 15;

    // ---- stage weights: global [k][n] f32 -> LDS [n][k] bf16 ----
    for (int idx = tid; idx < 128 * NCOLS; idx += 256) {
        int k = idx / NCOLS, n = idx % NCOLS;
        Blds[n * KP + k] = f32_to_bf16(W[idx]);
    }
    __syncthreads();

    // ---- WMMA main loop ----
    v8f acc[NT_T] = {};
    const float* Arow = A + (size_t)(blockIdx.x * 128 + wave * 16 + l16) * 128;
    const int kb = half * 8;                 // A-fragment K sub-offset per half

    for (int kk = 0; kk < 128; kk += 32) {
        // A fragment: lane holds row m = l16, K pairs per ISA 16-bit A layout
        float4 f0 = *(const float4*)(Arow + kk + kb);
        float4 f1 = *(const float4*)(Arow + kk + kb + 4);
        float4 f2 = *(const float4*)(Arow + kk + kb + 16);
        float4 f3 = *(const float4*)(Arow + kk + kb + 20);
        union { v16bf v; uint32_t u[8]; } a;
        a.u[0] = pack_bf16x2(f0.x, f0.y); a.u[1] = pack_bf16x2(f0.z, f0.w);
        a.u[2] = pack_bf16x2(f1.x, f1.y); a.u[3] = pack_bf16x2(f1.z, f1.w);
        a.u[4] = pack_bf16x2(f2.x, f2.y); a.u[5] = pack_bf16x2(f2.z, f2.w);
        a.u[6] = pack_bf16x2(f3.x, f3.y); a.u[7] = pack_bf16x2(f3.z, f3.w);

        #pragma unroll
        for (int t = 0; t < NT_T; ++t) {
            // B fragment: lane = column n, K order matches A (8h+0..7, 16+8h+0..7)
            const uint16_t* bp = &Blds[(t * 16 + l16) * KP + kk + half * 8];
            union { v16bf v; uint32_t u[8]; } b;
            #pragma unroll
            for (int v = 0; v < 8; ++v)
                b.u[v] = *(const uint32_t*)(bp + 2 * v + (v >= 4 ? 8 : 0));
            acc[t] = __builtin_amdgcn_wmma_f32_16x16x32_bf16(
                         false, a.v, false, b.v, (short)0, acc[t], false, false);
        }
    }

    // ---- epilogue: bias / relu / store / fused attention dots ----
    const size_t row0 = (size_t)blockIdx.x * 128;
    float es_r[8] = {0,0,0,0,0,0,0,0};
    float ed_r[8] = {0,0,0,0,0,0,0,0};
    const bool attn = (a_src != nullptr);

    #pragma unroll
    for (int t = 0; t < NT_T; ++t) {
        const int n = t * 16 + l16;
        const float bi = bias ? bias[n] : 0.0f;
        const float as = attn ? a_src[n] : 0.0f;
        const float ad = attn ? a_dst[n] : 0.0f;
        #pragma unroll
        for (int r = 0; r < 8; ++r) {
            float c = acc[t][r] + bi;
            if (do_relu) c = fmaxf(c, 0.0f);
            const int m_local = wave * 16 + r + half * 8;
            out[(row0 + m_local) * NCOLS + n] = c;
            if (attn) { es_r[r] += c * as; ed_r[r] += c * ad; }
        }
    }
    if (attn) {
        // reduce each row's partial dot across the 16 lanes of this half
        #pragma unroll
        for (int r = 0; r < 8; ++r) {
            float e1 = es_r[r], e2 = ed_r[r];
            #pragma unroll
            for (int off = 1; off < 16; off <<= 1) {
                e1 += __shfl_xor(e1, off, 32);
                e2 += __shfl_xor(e2, off, 32);
            }
            if (l16 == 0) {
                const int m_local = wave * 16 + r + half * 8;
                es[row0 + m_local] = e1;
                ed[row0 + m_local] = e2;
            }
        }
    }
}

// ---------------------------------------------------------------------------
// GAT softmax + aggregation. dst is implicitly node = edge/16 (edges are
// dst-sorted by construction) + one self loop. One wave per node:
// lanes 0..15 = the 16 in-edges, lane 16 = self loop.
// out[d] = relu( sum_j alpha_j * xw[src_j]  + bias )
// The j-loop is fully unrolled so all 17 row-gathers are in flight at once
// (served mostly from the 192MB L2: the whole xw buffer is L2-resident).
// ---------------------------------------------------------------------------
__global__ __launch_bounds__(256)
void gat_aggregate(const float* __restrict__ xw,     // [NT,128]
                   const float* __restrict__ es,     // [NT]
                   const float* __restrict__ ed,     // [NT]
                   const int*   __restrict__ src_s,  // supports src [80000*16]
                   const int*   __restrict__ src_q,  // queries  src [80000*16]
                   const float* __restrict__ bias,   // [128]
                   float* __restrict__ out)          // [NT,128]
{
    const int node = blockIdx.x * 8 + (threadIdx.x >> 5);
    const int lane = threadIdx.x & 31;

    int s = node;                                    // self loop / dummy
    if (lane < 16) {
        s = (node < NS_NODES)
              ? src_s[node * DEG + lane]
              : src_q[(node - NS_NODES) * DEG + lane] + NS_NODES;
    }
    if (lane <= 16)                                   // warm L2/L0 for gathers
        __builtin_prefetch(xw + (size_t)s * H_DIM, 0, 0);

    const float edn = ed[node];
    float e = -INFINITY;
    if (lane <= 16) {
        e = es[s] + edn;
        e = (e > 0.0f) ? e : NEG_SLOPE * e;           // leaky relu
    }
    float m = e;
    #pragma unroll
    for (int off = 16; off; off >>= 1) m = fmaxf(m, __shfl_xor(m, off, 32));
    float ex = (lane <= 16) ? __expf(e - m) : 0.0f;
    float den = ex;
    #pragma unroll
    for (int off = 16; off; off >>= 1) den += __shfl_xor(den, off, 32);
    const float alpha = ex / den;

    // broadcast all 17 (alpha, src) pairs, issue all 17 gathers, then FMA
    float aj[17];
    int   sj[17];
    #pragma unroll
    for (int j = 0; j <= 16; ++j) {
        aj[j] = __shfl(alpha, j, 32);
        sj[j] = __shfl(s, j, 32);
    }
    float4 v[17];
    #pragma unroll
    for (int j = 0; j <= 16; ++j)
        v[j] = *(const float4*)(xw + (size_t)sj[j] * H_DIM + lane * 4);

    float4 acc = {0.0f, 0.0f, 0.0f, 0.0f};
    #pragma unroll
    for (int j = 0; j <= 16; ++j) {
        acc.x += aj[j] * v[j].x; acc.y += aj[j] * v[j].y;
        acc.z += aj[j] * v[j].z; acc.w += aj[j] * v[j].w;
    }

    const float4 b4 = *(const float4*)(bias + lane * 4);
    acc.x = fmaxf(acc.x + b4.x, 0.0f);
    acc.y = fmaxf(acc.y + b4.y, 0.0f);
    acc.z = fmaxf(acc.z + b4.z, 0.0f);
    acc.w = fmaxf(acc.w + b4.w, 0.0f);
    *(float4*)(out + (size_t)node * H_DIM + lane * 4) = acc;
}

// ---------------------------------------------------------------------------
// Global mean pool: graph g covers nodes [200g, 200g+200) (concat layout makes
// query graphs contiguous too). One block per graph, 4 chunks x 64 features.
// ---------------------------------------------------------------------------
__global__ __launch_bounds__(256)
void pool_mean(const float* __restrict__ m2,   // [NT,64]
               float* __restrict__ gemb)       // [800,64]
{
    __shared__ float red[256];
    const int g = blockIdx.x;
    const int f = threadIdx.x & 63;
    const int chunk = threadIdx.x >> 6;
    const size_t base = (size_t)g * NPG;
    float sum = 0.0f;
    for (int i = chunk * 50; i < chunk * 50 + 50; ++i)
        sum += m2[(base + i) * OUT_DIM + f];
    red[threadIdx.x] = sum;
    __syncthreads();
    if (chunk == 0) {
        sum = red[f] + red[64 + f] + red[128 + f] + red[192 + f];
        gemb[g * OUT_DIM + f] = sum * (1.0f / (float)NPG);
    }
}

// ---------------------------------------------------------------------------
// Class prototypes per episode: mean of supports by label.
// Grid: 16 episodes, 64 threads (one per feature).
// ---------------------------------------------------------------------------
__global__ __launch_bounds__(64)
void make_proto(const float* __restrict__ gemb,   // support embeds [400,64]
                const int*   __restrict__ y,      // [400] labels 0..4
                float* __restrict__ proto)        // [16,5,64]
{
    const int b = blockIdx.x;
    const int f = threadIdx.x;
    float acc[N_WAY] = {0,0,0,0,0};
    float cnt[N_WAY] = {0,0,0,0,0};
    for (int k = 0; k < N_WAY * K_SHOT; ++k) {
        const int   c = y[b * N_WAY * K_SHOT + k];
        const float v = gemb[(b * N_WAY * K_SHOT + k) * OUT_DIM + f];
        #pragma unroll
        for (int c2 = 0; c2 < N_WAY; ++c2) {
            acc[c2] += (c2 == c) ? v : 0.0f;
            cnt[c2] += (c2 == c) ? 1.0f : 0.0f;
        }
    }
    #pragma unroll
    for (int c2 = 0; c2 < N_WAY; ++c2)
        proto[(b * N_WAY + c2) * OUT_DIM + f] = acc[c2] / fmaxf(cnt[c2], 1.0f);
}

// ---------------------------------------------------------------------------
// Final expansion: rep_q (repeat_interleave over classes) ++ rep_p (tile).
// row = (b*25 + j)*5 + n ; first half = emb_q[b,j], second half = proto[b,n].
// ---------------------------------------------------------------------------
__global__ __launch_bounds__(256)
void expand_out(const float* __restrict__ gemb,   // [800,64]; queries at +400
                const float* __restrict__ proto,  // [16,5,64]
                float* __restrict__ outp)
{
    const int idx = blockIdx.x * 256 + threadIdx.x;     // over 2000*64
    if (idx >= 2000 * OUT_DIM) return;
    const int f   = idx & 63;
    const int row = idx >> 6;
    const int n   = row % N_WAY;
    const int t   = row / N_WAY;                        // = b*25 + j
    outp[idx] = gemb[(400 + t) * OUT_DIM + f];
    outp[2000 * OUT_DIM + idx] = proto[((t / 25) * N_WAY + n) * OUT_DIM + f];
}

// ---------------------------------------------------------------------------
// Host orchestration
// ---------------------------------------------------------------------------
extern "C" void kernel_launch(void* const* d_in, const int* in_sizes, int n_in,
                              void* d_out, int out_size, void* d_ws, size_t ws_size,
                              hipStream_t stream) {
    const float* sup_x = (const float*)d_in[0];
    const float* qry_x = (const float*)d_in[1];
    const int*   sup_src = (const int*)d_in[2];   // edge_index[0] = src (first E)
    const int*   qry_src = (const int*)d_in[3];
    const int*   sup_y   = (const int*)d_in[6];
    const float* W1  = (const float*)d_in[7];
    const float* a1s = (const float*)d_in[8];
    const float* a1d = (const float*)d_in[9];
    const float* b1  = (const float*)d_in[10];
    const float* W2  = (const float*)d_in[11];
    const float* a2s = (const float*)d_in[12];
    const float* a2d = (const float*)d_in[13];
    const float* b2  = (const float*)d_in[14];
    const float* Wm1 = (const float*)d_in[15];
    const float* bm1 = (const float*)d_in[16];
    const float* Wm2 = (const float*)d_in[17];
    const float* bm2 = (const float*)d_in[18];

    // workspace carve-up (ping-pong NTx128 buffers + scalars)
    float* buf0  = (float*)d_ws;
    float* buf1  = buf0 + (size_t)NT_NODES * H_DIM;
    float* es    = buf1 + (size_t)NT_NODES * H_DIM;
    float* ed    = es + NT_NODES;
    float* gemb  = ed + NT_NODES;              // [800,64]
    float* proto = gemb + N_GRAPHS * OUT_DIM;  // [16,5,64]

    const dim3 blk(256);

    // ---- layer 1: xw = x @ W1, fused es/ed ----
    gemm_bf16_wmma<128><<<NS_NODES / 128, blk, 0, stream>>>(
        sup_x, W1, nullptr, a1s, a1d, buf0, es, ed, 0);
    gemm_bf16_wmma<128><<<NQ_NODES / 128, blk, 0, stream>>>(
        qry_x, W1, nullptr, a1s, a1d,
        buf0 + (size_t)NS_NODES * H_DIM, es + NS_NODES, ed + NS_NODES, 0);
    gat_aggregate<<<NT_NODES / 8, blk, 0, stream>>>(
        buf0, es, ed, sup_src, qry_src, b1, buf1);

    // ---- layer 2 ----
    gemm_bf16_wmma<128><<<NT_NODES / 128, blk, 0, stream>>>(
        buf1, W2, nullptr, a2s, a2d, buf0, es, ed, 0);
    gat_aggregate<<<NT_NODES / 8, blk, 0, stream>>>(
        buf0, es, ed, sup_src, qry_src, b2, buf1);

    // ---- MLP ----
    gemm_bf16_wmma<128><<<NT_NODES / 128, blk, 0, stream>>>(
        buf1, Wm1, bm1, nullptr, nullptr, buf0, nullptr, nullptr, 1);
    gemm_bf16_wmma<64><<<NT_NODES / 128, blk, 0, stream>>>(
        buf0, Wm2, bm2, nullptr, nullptr, buf1, nullptr, nullptr, 0);

    // ---- pool / prototypes / expand ----
    pool_mean<<<N_GRAPHS, blk, 0, stream>>>(buf1, gemb);
    make_proto<<<B_EP, dim3(64), 0, stream>>>(gemb, sup_y, proto);
    expand_out<<<(2000 * OUT_DIM + 255) / 256, blk, 0, stream>>>(
        gemb, proto, (float*)d_out);
}
